// HopfieldLayer_90494960926817
// MI455X (gfx1250) — compile-verified, matrix operand order checked by reading, not run on hardware
//
#include <hip/hip_runtime.h>
#include <stdint.h>

#define Bsz   4
#define Tsz   2048
#define Dsz   1024
#define NTILQ (Tsz / 16)          // 128 query tiles per batch
#define SCALE 0.03125f            // 1024^-0.5
#define SROW  2052                // padded score-row stride (floats); 8208B, 16B-aligned
#define NROWS (Bsz * Tsz)         // 8192 global rows

typedef __attribute__((ext_vector_type(16))) __bf16   v16bf;
typedef __attribute__((ext_vector_type(8)))  float    v8f;
typedef __attribute__((ext_vector_type(4)))  uint32_t v4u;
typedef __attribute__((ext_vector_type(2)))  uint32_t v2u;

__device__ __forceinline__ uint16_t f32_to_bf16(float f) {
  uint32_t u = __float_as_uint(f);
  u += 0x7FFFu + ((u >> 16) & 1u);   // round to nearest even
  return (uint16_t)(u >> 16);
}

union TileU { v4u u[2]; v16bf v; };

// Load one operand-row slice for v_wmma_f32_16x16x32_bf16.
// 16-bit A-matrix 16x32 layout: lane L -> row M = L%16, kb = (L/16)*8,
// vector elems 0..7  = K = kb+0 .. kb+7   (VGPR0..3)
// vector elems 8..15 = K = kb+16 .. kb+23 (VGPR4..7)
// B operand (32x16) mirrors this with N = L%16; both operands here come from
// row-major matrices whose K dimension is contiguous, so one loader serves both.
__device__ __forceinline__ v16bf load_tile_row(const uint16_t* row, int d0, int lane) {
  const int kb = (lane >> 4) << 3;
  TileU t;
  t.u[0] = *(const v4u*)(row + d0 + kb);
  t.u[1] = *(const v4u*)(row + d0 + kb + 16);
  return t.v;
}

// ---------------- Kernel 1: H fp32 -> bf16 ----------------
__global__ void k_convert_h(const float* __restrict__ H, uint16_t* __restrict__ hb) {
  size_t i = ((size_t)blockIdx.x * blockDim.x + threadIdx.x) * 4;
  float4 f = *(const float4*)(H + i);
  v2u p;
  p.x = (uint32_t)f32_to_bf16(f.x) | ((uint32_t)f32_to_bf16(f.y) << 16);
  p.y = (uint32_t)f32_to_bf16(f.z) | ((uint32_t)f32_to_bf16(f.w) << 16);
  *(v2u*)(hb + i) = p;
}

// ---------------- Kernel 2: W fp32 [D,E] -> bf16 transposed Wt[E,D] ----------------
__global__ void k_transpose_w(const float* __restrict__ W, uint16_t* __restrict__ wt) {
  __shared__ float tile[16][17];
  const int e0 = blockIdx.x * 16, d0 = blockIdx.y * 16;
  const int tx = threadIdx.x, ty = threadIdx.y;
  tile[ty][tx] = W[(size_t)(d0 + ty) * Dsz + e0 + tx];
  __syncthreads();
  wt[(size_t)(e0 + ty) * Dsz + d0 + tx] = f32_to_bf16(tile[tx][ty]);
}

// ---------------- Kernel 3: Ysem = (H @ W) * scale via bf16 WMMA ----------------
// 16x64 per wave: one A load feeds 4 WMMAs (2.5 b128 loads/WMMA). Compact
// register form (~62 VGPRs -> max occupancy, no spills); within-wave load
// latency is hidden across waves. (An explicitly double-buffered variant was
// tried and spilled to scratch in the hot loop -- strictly worse.)
__global__ void k_semantic(const uint16_t* __restrict__ hb,
                           const uint16_t* __restrict__ wt,
                           float* __restrict__ ysem) {
  const int lane = threadIdx.x & 31;
  const int wid  = blockIdx.x * (blockDim.x >> 5) + (threadIdx.x >> 5);
  const int tileM = wid >> 4;                 // 512 row-tiles of 16
  const int tileN = wid & 15;                 // 16  col-groups of 64
  const int row0 = tileM * 16, col0 = tileN * 64;

  const uint16_t* arow = hb + (size_t)(row0 + (lane & 15)) * Dsz;
  const uint16_t* br0  = wt + (size_t)(col0 +  0 + (lane & 15)) * Dsz;
  const uint16_t* br1  = wt + (size_t)(col0 + 16 + (lane & 15)) * Dsz;
  const uint16_t* br2  = wt + (size_t)(col0 + 32 + (lane & 15)) * Dsz;
  const uint16_t* br3  = wt + (size_t)(col0 + 48 + (lane & 15)) * Dsz;

  v8f acc0 = {}, acc1 = {}, acc2 = {}, acc3 = {};
  for (int d0 = 0; d0 < Dsz; d0 += 32) {
    v16bf a  = load_tile_row(arow, d0, lane);
    v16bf b0 = load_tile_row(br0,  d0, lane);
    v16bf b1 = load_tile_row(br1,  d0, lane);
    v16bf b2 = load_tile_row(br2,  d0, lane);
    v16bf b3 = load_tile_row(br3,  d0, lane);
    acc0 = __builtin_amdgcn_wmma_f32_16x16x32_bf16(false, a, false, b0, (short)0, acc0, false, false);
    acc1 = __builtin_amdgcn_wmma_f32_16x16x32_bf16(false, a, false, b1, (short)0, acc1, false, false);
    acc2 = __builtin_amdgcn_wmma_f32_16x16x32_bf16(false, a, false, b2, (short)0, acc2, false, false);
    acc3 = __builtin_amdgcn_wmma_f32_16x16x32_bf16(false, a, false, b3, (short)0, acc3, false, false);
  }

  const int mh = (lane >> 4) << 3;
  const int nl = lane & 15;
#pragma unroll
  for (int r = 0; r < 8; ++r) {
    const size_t m = (size_t)(row0 + r + mh) * Dsz + col0 + nl;
    ysem[m +  0] = acc0[r] * SCALE;
    ysem[m + 16] = acc1[r] * SCALE;
    ysem[m + 32] = acc2[r] * SCALE;
    ysem[m + 48] = acc3[r] * SCALE;
  }
}

// ---------------- Kernel 4: fused episodic scores -> top-64 -> output ----------------
// One workgroup (16 waves) per (batch, 16-query tile). LDS holds the full
// 16 x 2048 fp32 causal score strip (131 KB of the 320 KB WGP LDS).
__global__ void k_episodic(const uint16_t* __restrict__ hb,
                           const float* __restrict__ H,
                           const float* __restrict__ ysem,
                           float* __restrict__ out) {
  extern __shared__ float smem[];
  float* scores = smem;                       // [16][SROW]
  float* topv   = smem + 16 * SROW;           // [16][64]
  int*   topi   = (int*)(topv + 16 * 64);     // [16][64]

  const int lane = threadIdx.x & 31;
  const int wave = threadIdx.x >> 5;          // 0..15
  const int qt   = blockIdx.x % NTILQ;
  const int b    = blockIdx.x / NTILQ;
  const int q0   = qt * 16;
  const uint16_t* hbB = hb + (size_t)b * Tsz * Dsz;

  // ---- Phase 1: WMMA scores, two key tiles per A load (3 b128 loads / WMMA),
  // software-pipelined one chunk deep (small live set: ~64 VGPRs, no spills).
  // Pair p covers key tiles {2p, 2p+1}; the second tile of the last odd pair is
  // clamped (computed, not stored) so the K-loop stays branch-free and EXEC
  // stays all-1s around every WMMA.
  const uint16_t* arow = hbB + (size_t)(q0 + (lane & 15)) * Dsz;
  const int npairs = (qt >> 1) + 1;
  for (int p = wave; p < npairs; p += 16) {
    const int kt0 = 2 * p;
    const int kt1 = kt0 + 1;
    const bool have1 = (kt1 <= qt);
    const int kt1c = have1 ? kt1 : kt0;       // clamp: safe dummy compute
    const uint16_t* brow0 = hbB + (size_t)(kt0 * 16 + (lane & 15)) * Dsz;
    const uint16_t* brow1 = hbB + (size_t)(kt1c * 16 + (lane & 15)) * Dsz;

    v8f acc0 = {}, acc1 = {};
    v16bf a  = load_tile_row(arow,  0, lane);
    v16bf b0 = load_tile_row(brow0, 0, lane);
    v16bf b1 = load_tile_row(brow1, 0, lane);
    for (int d0 = 0; d0 < Dsz - 32; d0 += 32) {
      const int dn = d0 + 32;
      v16bf an  = load_tile_row(arow,  dn, lane);
      v16bf bn0 = load_tile_row(brow0, dn, lane);
      v16bf bn1 = load_tile_row(brow1, dn, lane);
      acc0 = __builtin_amdgcn_wmma_f32_16x16x32_bf16(false, a, false, b0, (short)0, acc0, false, false);
      acc1 = __builtin_amdgcn_wmma_f32_16x16x32_bf16(false, a, false, b1, (short)0, acc1, false, false);
      a = an; b0 = bn0; b1 = bn1;
    }
    acc0 = __builtin_amdgcn_wmma_f32_16x16x32_bf16(false, a, false, b0, (short)0, acc0, false, false);
    acc1 = __builtin_amdgcn_wmma_f32_16x16x32_bf16(false, a, false, b1, (short)0, acc1, false, false);

    const int nl = lane & 15;
    const int mh = (lane >> 4) << 3;
    {
      const int j = kt0 * 16 + nl;
#pragma unroll
      for (int r = 0; r < 8; ++r) {
        int m = r + mh;
        scores[m * SROW + j] = (j < q0 + m) ? acc0[r] * SCALE : -__builtin_inff();
      }
    }
    if (have1) {
      const int j = kt1 * 16 + nl;
#pragma unroll
      for (int r = 0; r < 8; ++r) {
        int m = r + mh;
        scores[m * SROW + j] = (j < q0 + m) ? acc1[r] * SCALE : -__builtin_inff();
      }
    }
  }
  __syncthreads();

  // ---- Phase 2: per-row top-64 (wave w owns row w); destructive wave argmax.
  // Scan bound is block-uniform: every computed entry with j >= i is already
  // -inf, so scanning the full (qt+1)*16 strip with ds_load_b128 is safe,
  // divergence-free, and 4x fewer DS ops than scalar.
  {
    const int m = wave;
    float* row = scores + m * SROW;
    const float4* row4 = (const float4*)row;
    const int nv4 = (qt + 1) * 4;             // strip length in float4 units
    for (int it = 0; it < 64; ++it) {
      float bestv = -__builtin_inff();
      int   bestj = -1;
      for (int j4 = lane; j4 < nv4; j4 += 32) {
        float4 v = row4[j4];
        const int jb = j4 * 4;
        if (v.x > bestv) { bestv = v.x; bestj = jb; }
        if (v.y > bestv) { bestv = v.y; bestj = jb + 1; }
        if (v.z > bestv) { bestv = v.z; bestj = jb + 2; }
        if (v.w > bestv) { bestv = v.w; bestj = jb + 3; }
      }
#pragma unroll
      for (int off = 16; off > 0; off >>= 1) {
        float ov = __shfl_xor(bestv, off, 32);
        int   oj = __shfl_xor(bestj, off, 32);
        if (ov > bestv || (ov == bestv && oj < bestj)) { bestv = ov; bestj = oj; }
      }
      if (lane == 0) {
        if (bestj >= 0 && bestv > -__builtin_inff()) {
          topv[m * 64 + it] = bestv;
          topi[m * 64 + it] = bestj;
          row[bestj] = -__builtin_inff();     // LDS in-order per wave: seen next pass
        } else {
          topv[m * 64 + it] = 0.0f;
          topi[m * 64 + it] = 0;
        }
      }
    }
  }
  __syncthreads();

  // ---- Phase 3: y_ep gather (L2-resident H), gate, blend, store
  {
    const int m = wave;
    const size_t grow = ((size_t)b * Tsz + q0 + m) * Dsz;
    float4 acc4[8];
#pragma unroll
    for (int t = 0; t < 8; ++t) acc4[t] = make_float4(0.f, 0.f, 0.f, 0.f);

    const float* tv = topv + m * 64;
    const int*   ti = topi + m * 64;
    for (int s = 0; s < 64; ++s) {
      float v = tv[s];                        // wave-uniform
      if (s + 1 < 64) {                       // warm WGP$ for the next gather row
        __builtin_prefetch(H + ((size_t)b * Tsz + ti[s + 1]) * Dsz + lane * 4, 0, 1);
      }
      if (v == 0.0f) continue;
      const float4* hrow = (const float4*)(H + ((size_t)b * Tsz + ti[s]) * Dsz) + lane;
#pragma unroll
      for (int t = 0; t < 8; ++t) {
        float4 h = hrow[t * 32];
        acc4[t].x += v * h.x; acc4[t].y += v * h.y;
        acc4[t].z += v * h.z; acc4[t].w += v * h.w;
      }
    }

    const float4* ysp = (const float4*)(ysem + grow) + lane;
    float4 ysv[8];
    float dot = 0.f;
#pragma unroll
    for (int t = 0; t < 8; ++t) {
      ysv[t] = ysp[t * 32];
      dot += acc4[t].x * ysv[t].x + acc4[t].y * ysv[t].y +
             acc4[t].z * ysv[t].z + acc4[t].w * ysv[t].w;
    }
#pragma unroll
    for (int off = 16; off > 0; off >>= 1) dot += __shfl_xor(dot, off, 32);

    const float gate = 1.0f / (1.0f + __expf(-dot));
    const float gi   = 1.0f - gate;
    float4* op = (float4*)(out + grow) + lane;
#pragma unroll
    for (int t = 0; t < 8; ++t) {
      float4 o;
      o.x = gate * acc4[t].x + gi * ysv[t].x;
      o.y = gate * acc4[t].y + gi * ysv[t].y;
      o.z = gate * acc4[t].z + gi * ysv[t].z;
      o.w = gate * acc4[t].w + gi * ysv[t].w;
      op[t * 32] = o;
    }
  }
}

extern "C" void kernel_launch(void* const* d_in, const int* in_sizes, int n_in,
                              void* d_out, int out_size, void* d_ws, size_t ws_size,
                              hipStream_t stream) {
  const float* H = (const float*)d_in[0];    // [B,T,D] fp32
  const float* W = (const float*)d_in[1];    // [D,D]   fp32
  float* out = (float*)d_out;

  // Workspace layout: Hb bf16 (16MB) | Wt bf16 transposed (2MB) | Ysem f32 (32MB)
  uint16_t* hb = (uint16_t*)d_ws;
  uint16_t* wt = hb + (size_t)Bsz * Tsz * Dsz;
  float* ysem  = (float*)(wt + (size_t)Dsz * Dsz);

  // 1) H -> bf16 (float4 per thread)
  {
    const size_t n4 = (size_t)Bsz * Tsz * Dsz / 4;
    k_convert_h<<<(unsigned)(n4 / 256), 256, 0, stream>>>(H, hb);
  }
  // 2) W -> bf16 transposed
  k_transpose_w<<<dim3(Dsz / 16, Dsz / 16), dim3(16, 16), 0, stream>>>(W, wt);

  // 3) Ysem = scale * H @ W  (bf16 WMMA, 16x64 per wave, 8 waves/WG)
  {
    const int wave_tiles = (NROWS / 16) * (Dsz / 64);   // 8192
    k_semantic<<<wave_tiles / 8, 256, 0, stream>>>(hb, wt, ysem);
  }

  // 4) Fused episodic: scores (WMMA) -> top-64 -> gather -> gate -> out
  {
    const int smem_bytes = (16 * SROW + 16 * 64) * (int)sizeof(float)
                         + 16 * 64 * (int)sizeof(int);        // 139,520 B
    hipFuncSetAttribute((const void*)k_episodic,
                        hipFuncAttributeMaxDynamicSharedMemorySize, smem_bytes);
    k_episodic<<<Bsz * NTILQ, 512, smem_bytes, stream>>>(hb, H, ysem, out);
  }
}
